// QuantizedLinearLayer_51677046505998
// MI455X (gfx1250) — matrix operand order repaired and checked by reading, not compiled, not run on hardware
//
#include <hip/hip_runtime.h>
#include <hip/hip_bf16.h>
#include <stdint.h>

// ---------------------------------------------------------------------------
// Types
// ---------------------------------------------------------------------------
typedef __attribute__((ext_vector_type(16))) __bf16 v16bf;
typedef __attribute__((ext_vector_type(8)))  __bf16 v8bf;
typedef __attribute__((ext_vector_type(4)))  __bf16 v4bf;
typedef __attribute__((ext_vector_type(8)))  float  v8f;
typedef __attribute__((ext_vector_type(4)))  float  v4f;
typedef __attribute__((ext_vector_type(4)))  int    v4i;

struct AFragPair { v8bf lo, hi; };   // for bit_cast into v16bf

// ---------------------------------------------------------------------------
// Problem constants
// ---------------------------------------------------------------------------
#define OUT_F   11008
#define IN_F    4096
#define M_TOT   8192           // 4 * 2048
#define GROUP   128

#define BM      128            // block tile M (2 waves x 64)
#define BN      256            // block tile N (4 waves x 64)
#define BK      32
#define BKP     40             // padded LDS row stride (80 B): kills 4-way
                               // bank aliasing of the 64 B natural stride
#define KTILES  (IN_F / BK)    // 128

__constant__ float NF4_TAB[16] = {
    -1.0f, -0.6961928009986877f, -0.5250730514526367f, -0.39491748809814453f,
    -0.28444138169288635f, -0.18477343022823334f, -0.09105003625154495f, 0.0f,
    0.07958029955625534f, 0.16093020141124725f, 0.24611230194568634f,
    0.33791524171829224f, 0.44070982933044434f, 0.5626170039176941f,
    0.7229568362236023f, 1.0f};

// ---------------------------------------------------------------------------
// Kernel 1: NF4 dequant -> bf16 W[OUT_F][IN_F] (row-major, K contiguous)
// 8 weights per thread; group-of-128 scale is constant within the 8.
// ---------------------------------------------------------------------------
__global__ __launch_bounds__(256) void nf4_dequant_kernel(
    const int* __restrict__ q, const float* __restrict__ scale,
    __bf16* __restrict__ Wb) {
  __shared__ float tab[16];
  if (threadIdx.x < 16) tab[threadIdx.x] = NF4_TAB[threadIdx.x];
  __syncthreads();

  size_t base = ((size_t)blockIdx.x * 256 + threadIdx.x) * 8;
  float sc = scale[base >> 7];
  v4i q0 = *(const v4i*)(q + base);
  v4i q1 = *(const v4i*)(q + base + 4);

  v8bf w;
  w[0] = (__bf16)(tab[q0[0] & 15] * sc);
  w[1] = (__bf16)(tab[q0[1] & 15] * sc);
  w[2] = (__bf16)(tab[q0[2] & 15] * sc);
  w[3] = (__bf16)(tab[q0[3] & 15] * sc);
  w[4] = (__bf16)(tab[q1[0] & 15] * sc);
  w[5] = (__bf16)(tab[q1[1] & 15] * sc);
  w[6] = (__bf16)(tab[q1[2] & 15] * sc);
  w[7] = (__bf16)(tab[q1[3] & 15] * sc);
  *(v8bf*)(Wb + base) = w;
}

// ---------------------------------------------------------------------------
// Kernel 2: x fp32 -> bf16, 4 elements / thread
// ---------------------------------------------------------------------------
__global__ __launch_bounds__(256) void xconv_kernel(
    const float* __restrict__ x, __bf16* __restrict__ Xb) {
  size_t base = ((size_t)blockIdx.x * 256 + threadIdx.x) * 4;
  v4f v = *(const v4f*)(x + base);
  v4bf o;
  o[0] = (__bf16)v[0]; o[1] = (__bf16)v[1];
  o[2] = (__bf16)v[2]; o[3] = (__bf16)v[3];
  *(v4bf*)(Xb + base) = o;
}

// ---------------------------------------------------------------------------
// CDNA5 async copy: global -> LDS, 16B per lane, tracked by ASYNCcnt
// ---------------------------------------------------------------------------
__device__ __forceinline__ void async_copy_b128(uint32_t lds_off,
                                                const void* gptr) {
  asm volatile("global_load_async_to_lds_b128 %0, %1, off"
               :: "v"(lds_off), "v"((uint64_t)(uintptr_t)gptr)
               : "memory");
}

// ---------------------------------------------------------------------------
// Kernel 3: bf16 WMMA GEMM, 128x256 block tile, BK=32, double-buffered LDS.
// 8 waves as 2(M) x 4(N); wave tile 64x64 = 4x4 WMMA 16x16 accumulators.
// 16 WMMAs per 16 ds_load_b128 per wave per K-step (0.5 KB LDS read / WMMA).
// out[m][n] = sum_k Xb[m][k] * Wb[n][k] + bias[n]
// ---------------------------------------------------------------------------
__global__ __launch_bounds__(256) void nf4_gemm_kernel(
    const __bf16* __restrict__ Xb, const __bf16* __restrict__ Wb,
    const float* __restrict__ bias, float* __restrict__ out) {
  __shared__ __bf16 As[2][BM * BKP];   // 10 KB per buffer (padded rows)
  __shared__ __bf16 Bs[2][BN * BKP];   // 20 KB per buffer (padded rows)

  const int tid   = threadIdx.x;
  const int lane  = tid & 31;
  const int wave  = tid >> 5;
  const int l16   = lane & 15;
  const int lh    = lane >> 4;        // 0 or 1 (half-wave)
  const int waveM = wave >> 2;        // 0..1
  const int waveN = wave & 3;         // 0..3
  const int bM    = blockIdx.y * BM;
  const int bN    = blockIdx.x * BN;

  // A tile: 512 16B chunks -> 2/thread. B tile: 1024 chunks -> 4/thread.
  auto issue_tiles = [&](int buf, int kt) {
#pragma unroll
    for (int t = 0; t < 2; ++t) {
      int cid = tid + t * 256;            // 0..511
      int r   = cid >> 2;                 // row within A tile
      int c   = (cid & 3) << 3;           // k offset (8 bf16 = 16B)
      async_copy_b128((uint32_t)(uintptr_t)&As[buf][r * BKP + c],
                      Xb + (size_t)(bM + r) * IN_F + (size_t)kt * BK + c);
    }
#pragma unroll
    for (int t = 0; t < 4; ++t) {
      int cid = tid + t * 256;            // 0..1023
      int r   = cid >> 2;                 // row within B tile
      int c   = (cid & 3) << 3;
      async_copy_b128((uint32_t)(uintptr_t)&Bs[buf][r * BKP + c],
                      Wb + (size_t)(bN + r) * IN_F + (size_t)kt * BK + c);
    }
  };

  v8f zero = {};
  v8f acc[4][4];
#pragma unroll
  for (int m = 0; m < 4; ++m)
#pragma unroll
    for (int n = 0; n < 4; ++n) acc[m][n] = zero;

  issue_tiles(0, 0);

  for (int kt = 0; kt < KTILES; ++kt) {
    const int buf = kt & 1;
    if (kt + 1 < KTILES) {
      issue_tiles(buf ^ 1, kt + 1);                       // prefetch next
      asm volatile("s_wait_asynccnt 6" ::: "memory");     // current 6 done
    } else {
      asm volatile("s_wait_asynccnt 0" ::: "memory");
    }
    __syncthreads();   // make async-written LDS visible to all waves

    // A fragments: 16x32 bf16. lane<16: K 0..15; lane>=16: K 16..31
    // -> two contiguous 16B chunks at lh*16, +8.
    v16bf aF[4];
#pragma unroll
    for (int m = 0; m < 4; ++m) {
      const __bf16* p = &As[buf][(waveM * 64 + m * 16 + l16) * BKP + lh * 16];
      AFragPair fp;
      fp.lo = *(const v8bf*)p;
      fp.hi = *(const v8bf*)(p + 8);
      aF[m] = __builtin_bit_cast(v16bf, fp);
    }
    // B fragments: 32x16 bf16, col N = l16, K = lh*16 .. +15 contiguous 32B.
    v16bf bF[4];
#pragma unroll
    for (int n = 0; n < 4; ++n) {
      const __bf16* p = &Bs[buf][(waveN * 64 + n * 16 + l16) * BKP + lh * 16];
      bF[n] = *(const v16bf*)p;
    }

#pragma unroll
    for (int m = 0; m < 4; ++m)
#pragma unroll
      for (int n = 0; n < 4; ++n)
        acc[m][n] = __builtin_amdgcn_wmma_f32_16x16x32_bf16(
            false, aF[m], false, bF[n], (short)0, acc[m][n], false, false);

    __syncthreads();   // done reading buf before it is refilled next+1 iter
  }

  // Epilogue: C/D layout -> lane l16 = column N, VGPR v + 8*lh = row M.
#pragma unroll
  for (int n = 0; n < 4; ++n) {
    int col  = bN + waveN * 64 + n * 16 + l16;
    float bv = bias[col];
#pragma unroll
    for (int m = 0; m < 4; ++m) {
      int row0 = bM + waveM * 64 + m * 16 + lh * 8;
#pragma unroll
      for (int v = 0; v < 8; ++v)
        out[(size_t)(row0 + v) * OUT_F + col] = acc[m][n][v] + bv;
    }
  }
}

// ---------------------------------------------------------------------------
// Launch: inputs are x[f32], weight_quant[i32], weight_scale[f32], bias[f32]
// ws layout: Wb bf16 [OUT_F*IN_F] (90.2 MB) | Xb bf16 [M_TOT*IN_F] (67.1 MB)
// ---------------------------------------------------------------------------
extern "C" void kernel_launch(void* const* d_in, const int* in_sizes, int n_in,
                              void* d_out, int out_size, void* d_ws,
                              size_t ws_size, hipStream_t stream) {
  const float* x      = (const float*)d_in[0];
  const int*   wq     = (const int*)d_in[1];
  const float* wscale = (const float*)d_in[2];
  const float* bias   = (const float*)d_in[3];
  float*       out    = (float*)d_out;

  __bf16* Wb = (__bf16*)d_ws;
  __bf16* Xb = Wb + (size_t)OUT_F * IN_F;

  // 45,088,768 weights / 8 per thread / 256 = 22016 blocks (exact)
  nf4_dequant_kernel<<<22016, 256, 0, stream>>>(wq, wscale, Wb);
  // 33,554,432 x elems / 4 per thread / 256 = 32768 blocks (exact)
  xconv_kernel<<<32768, 256, 0, stream>>>(x, Xb);
  // 43 N-tiles x 64 M-tiles
  nf4_gemm_kernel<<<dim3(OUT_F / BN, M_TOT / BM), 256, 0, stream>>>(
      Xb, Wb, bias, out);
}
